// SAGENet_1477468750396
// MI455X (gfx1250) — matrix operand order repaired and checked by reading, not compiled, not run on hardware
//
#include <hip/hip_runtime.h>
#include <hip/hip_bf16.h>

#define N_NODES 50000
#define N_EDGES 600000
#define N_LABEL 100000
#define IN_C 128
#define HID_C 128
#define OUT_C 64

typedef __attribute__((ext_vector_type(16))) __bf16 v16bf;
typedef __attribute__((ext_vector_type(8)))  float  v8f;

__device__ __forceinline__ __bf16 f2bf(float f) { return (__bf16)f; }

// ---------------------------------------------------------------- utilities
__global__ void zero_kernel(float* __restrict__ p, int n) {
    int i = blockIdx.x * blockDim.x + threadIdx.x;
    int stride = gridDim.x * blockDim.x;
    for (; i < n; i += stride) p[i] = 0.0f;
}

// degree: one atomic per edge (deg fits in L2, atomics resolve there)
__global__ void deg_kernel(const int* __restrict__ dst, float* __restrict__ deg) {
    int e = blockIdx.x * blockDim.x + threadIdx.x;
    if (e < N_EDGES) atomicAdd(&deg[dst[e]], 1.0f);
}

// scatter-add: 32 lanes per edge, 4 channels per lane (128 channels).
// float4 gather is coalesced per edge-group; atomics land in L2 (agg = 25.6MB).
__global__ void scatter_kernel(const float* __restrict__ feat,
                               const int* __restrict__ src,
                               const int* __restrict__ dst,
                               float* __restrict__ agg) {
    int idx = blockIdx.x * blockDim.x + threadIdx.x;
    int e = idx >> 5;
    if (e >= N_EDGES) return;
    int c = (idx & 31) << 2;
    int s = src[e];
    int d = dst[e];
    const float4 v = *(const float4*)(feat + (size_t)s * IN_C + c);
    float* o = agg + (size_t)d * IN_C + c;
    atomicAdd(o + 0, v.x);
    atomicAdd(o + 1, v.y);
    atomicAdd(o + 2, v.z);
    atomicAdd(o + 3, v.w);
}

// ---------------------------------------------------------------- fused SAGE layer
// out[16x16 tile] = (agg/deg) @ Wl^T + x @ Wr^T + b, optional ReLU.
// One wave per tile; K=128 -> 4 chunks of 32; 2 WMMAs per chunk chained on acc.
//
// A (16x32 bf16) layout (ISA 7.12.2): lane L holds row L&15; VGPR j (j<4) holds
//   K = kbase+2j, kbase+2j+1 with kbase = (L>>4)*8; VGPRs 4-7 the same +16.
//   => lane's 16 elements are K in [kbase, kbase+8) ++ [kbase+16, kbase+24).
// B (32x16 bf16): lane L holds column L&15; same K order as A:
//   [kbase, kbase+8) ++ [kbase+16, kbase+24). B[k][n] = W[n][k] => row of W.
// C/D (16x16 f32): VGPR v -> row v + 8*(L>>4), col L&15.
template <int COUT, bool RELU>
__global__ __launch_bounds__(32)
void sage_gemm_kernel(const float* __restrict__ agg,
                      const float* __restrict__ deg,
                      const float* __restrict__ xin,
                      const float* __restrict__ wl,    // [COUT][IN_C] row-major
                      const float* __restrict__ wr,    // [COUT][IN_C] row-major
                      const float* __restrict__ bias,  // [COUT]
                      float* __restrict__ out) {       // [N_NODES][COUT]
    const int lane = threadIdx.x;           // 0..31, EXEC all ones
    const int tm   = blockIdx.x;            // row tile  (N_NODES/16 = 3125)
    const int tn   = blockIdx.y;            // col tile  (COUT/16)
    const int rowA = tm * 16 + (lane & 15);
    const int ka   = (lane >> 4) * 8;
    const int colB = tn * 16 + (lane & 15);
    const int kb   = (lane >> 4) * 8;

    const float dscale = 1.0f / fmaxf(deg[rowA], 1.0f);
    const float* aggRow = agg + (size_t)rowA * IN_C;
    const float* xRow   = xin + (size_t)rowA * IN_C;
    const float* wlRow  = wl  + (size_t)colB * IN_C;
    const float* wrRow  = wr  + (size_t)colB * IN_C;

    v8f acc = {};
#pragma unroll
    for (int kc = 0; kc < IN_C; kc += 32) {
        v16bf am, ax, bl, br;
#pragma unroll
        for (int i = 0; i < 8; ++i) {
            am[i]     = f2bf(aggRow[kc + ka + i] * dscale);
            am[8 + i] = f2bf(aggRow[kc + 16 + ka + i] * dscale);
            ax[i]     = f2bf(xRow[kc + ka + i]);
            ax[8 + i] = f2bf(xRow[kc + 16 + ka + i]);
        }
#pragma unroll
        for (int i = 0; i < 16; ++i) {
            bl[i] = f2bf(wlRow[kc + kb + i + (i & 8)]);
            br[i] = f2bf(wrRow[kc + kb + i + (i & 8)]);
        }
        acc = __builtin_amdgcn_wmma_f32_16x16x32_bf16(false, am, false, bl,
                                                      (short)0, acc, false, false);
        acc = __builtin_amdgcn_wmma_f32_16x16x32_bf16(false, ax, false, br,
                                                      (short)0, acc, false, false);
    }

    const float bv    = bias[colB];
    const int   rbase = tm * 16 + 8 * (lane >> 4);
#pragma unroll
    for (int v = 0; v < 8; ++v) {
        float r = acc[v] + bv;
        if (RELU) r = fmaxf(r, 0.0f);
        out[(size_t)(rbase + v) * COUT + colB] = r;
    }
}

// ---------------------------------------------------------------- decode
// one wave per label edge; 2 channels/lane over OUT_C=64; shfl_xor reduce.
__global__ void decode_kernel(const float* __restrict__ z,
                              const int* __restrict__ ea,
                              const int* __restrict__ eb,
                              float* __restrict__ out) {
    int idx  = blockIdx.x * blockDim.x + threadIdx.x;
    int e    = idx >> 5;
    int lane = idx & 31;
    if (e >= N_LABEL) return;
    int a = ea[e];
    int b = eb[e];
    const float* za = z + (size_t)a * OUT_C;
    const float* zb = z + (size_t)b * OUT_C;
    float p = za[lane] * zb[lane] + za[lane + 32] * zb[lane + 32];
#pragma unroll
    for (int off = 16; off > 0; off >>= 1) p += __shfl_xor(p, off, 32);
    if (lane == 0) out[e] = p;
}

// ---------------------------------------------------------------- launch
extern "C" void kernel_launch(void* const* d_in, const int* in_sizes, int n_in,
                              void* d_out, int out_size, void* d_ws, size_t ws_size,
                              hipStream_t stream) {
    const float* x   = (const float*)d_in[0];
    const int*   ei  = (const int*)d_in[1];  // [2][N_EDGES]: src row, dst row
    const int*   eli = (const int*)d_in[2];  // [2][N_LABEL]
    const float* w1l = (const float*)d_in[3];
    const float* b1l = (const float*)d_in[4];
    const float* w1r = (const float*)d_in[5];
    const float* w2l = (const float*)d_in[6];
    const float* b2l = (const float*)d_in[7];
    const float* w2r = (const float*)d_in[8];
    float*       out = (float*)d_out;

    // workspace layout (floats): deg | agg | h | z   (~64.2 MB total)
    float* deg = (float*)d_ws;
    float* agg = deg + 50048;                       // padded for alignment
    float* h   = agg + (size_t)N_NODES * IN_C;
    float* z   = h   + (size_t)N_NODES * HID_C;

    const int* src = ei;
    const int* dst = ei + N_EDGES;

    const int scatter_blocks = (N_EDGES * 32 + 255) / 256;

    // layer 1
    zero_kernel<<<4096, 256, 0, stream>>>(deg, 50048 + N_NODES * IN_C);
    deg_kernel<<<(N_EDGES + 255) / 256, 256, 0, stream>>>(dst, deg);
    scatter_kernel<<<scatter_blocks, 256, 0, stream>>>(x, src, dst, agg);
    sage_gemm_kernel<HID_C, true>
        <<<dim3(N_NODES / 16, HID_C / 16), 32, 0, stream>>>(agg, deg, x, w1l, w1r, b1l, h);

    // layer 2 (reuse agg, re-zero)
    zero_kernel<<<4096, 256, 0, stream>>>(agg, N_NODES * IN_C);
    scatter_kernel<<<scatter_blocks, 256, 0, stream>>>(h, src, dst, agg);
    sage_gemm_kernel<OUT_C, false>
        <<<dim3(N_NODES / 16, OUT_C / 16), 32, 0, stream>>>(agg, deg, h, w2l, w2r, b2l, z);

    // decode
    decode_kernel<<<(N_LABEL * 32 + 255) / 256, 256, 0, stream>>>(z, eli, eli + N_LABEL, out);
}
